// ResNet_5686536700365
// MI455X (gfx1250) — 8-bit matrix operand order repaired and checked by reading, compiled, not run on hardware
//
#include <hip/hip_runtime.h>


#define RESOLUTION 32
#define BATCH 512

typedef __attribute__((ext_vector_type(8))) int v8i;

static inline int cdiv_h(int a, int b) { return (a + b - 1) / b; }

__global__ void prep_w_kernel(const float* __restrict__ w, signed char* __restrict__ wp,
                              int CIN, int COUT, int KS, int KPAD) {
  int i = blockIdx.x * blockDim.x + threadIdx.x;
  int total = KPAD * COUT;
  if (i >= total) return;
  int k = i / COUT, n = i - (i / COUT) * COUT;
  int KTOT = CIN * KS * KS;
  signed char v = 0;
  if (k < KTOT) {
    int t = k / CIN, c = k - t * CIN;
    int ky = t / KS, kx = t - ky * KS;
    float wv = w[((size_t)(n * CIN + c) * KS + ky) * KS + kx];
    v = (wv >= 0.0f) ? (signed char)1 : (signed char)-1;
  }
  wp[(size_t)(k >> 2) * COUT * 4 + n * 4 + (k & 3)] = v;
}

template <int CIN, int COUT, int STRIDE, int KS, bool PG, int HIN, int WIN,
          bool EMIT, bool HASRES>
__global__ void __launch_bounds__(256)
pg_conv_wmma(const signed char* __restrict__ xb1, const signed char* __restrict__ xb2,
             const signed char* __restrict__ wp,
             const float* __restrict__ bns, const float* __restrict__ bnb,
             const float* __restrict__ th, const float* __restrict__ res,
             float* __restrict__ out,
             signed char* __restrict__ ob1, signed char* __restrict__ ob2) {
  constexpr int HO = HIN / STRIDE, WO = WIN / STRIDE;
  constexpr int NPIX = BATCH * HO * WO;
  constexpr int KTOT = CIN * KS * KS;
  constexpr int KCH = (KTOT + 63) / 64;
  constexpr int NT = COUT / 16;
  constexpr int PADV = (KS == 3) ? 1 : 0;
  constexpr int PAR = (NT == 1) ? 2 : 1;

  const int wave = (int)((blockIdx.x * blockDim.x + threadIdx.x) >> 5);
  const int lane = threadIdx.x & 31;
  const int mtile = wave;
  if (mtile * 16 >= NPIX) return;

  const int row = lane & 15, grp = lane >> 4;
  const int p = mtile * 16 + row;
  const int bb = p / (HO * WO);
  const int rp = p - bb * (HO * WO);
  const int oy = rp / WO, ox = rp - (rp / WO) * WO;

  const int sbase = ((bb * HIN + (oy * STRIDE - PADV)) * WIN + (ox * STRIDE - PADV)) * CIN
                    + (grp << 3);
  const signed char* pA1 = xb1 + (long)sbase;
  const signed char* pA2 = PG ? (xb2 + (long)sbase) : pA1;

  unsigned tmask = 0u;
#pragma unroll
  for (int t = 0; t < KS * KS; ++t) {
    int ky = t / KS, kx = t - (t / KS) * KS;
    int iy = oy * STRIDE + ky - PADV, ix = ox * STRIDE + kx - PADV;
    unsigned bit = (iy >= 0 && iy < HIN && ix >= 0 && ix < WIN) ? 1u : 0u;
    tmask |= bit << t;
  }

  const int* wp32 = (const int*)wp;
  const int wOff = (grp << 1) * COUT + (lane & 15);

  v8i acc1[NT][PAR], acc2[NT][PAR];
#pragma unroll
  for (int nt = 0; nt < NT; ++nt)
#pragma unroll
    for (int pr = 0; pr < PAR; ++pr) { acc1[nt][pr] = (v8i){}; acc2[nt][pr] = (v8i){}; }

#pragma unroll
  for (int kc = 0; kc < KCH; ++kc) {
    v8i a1, a2 = {};
#pragma unroll
    for (int v = 0; v < 8; ++v) {
      const int kb0 = kc * 64 + ((v >> 1) << 4) + ((v & 1) << 2);
      const int t0 = kb0 / CIN, c0 = kb0 - t0 * CIN;
      const bool cross = (c0 + 8 >= CIN);
      const int ky0 = t0 / KS, kx0 = t0 - ky0 * KS;
      const int C0 = (ky0 * WIN + kx0) * CIN + c0;
      const bool rowx = cross && (kx0 == KS - 1);
      const int DELTA = (WIN - KS) * CIN;
      int sh = t0 + ((cross) ? grp : 0);
      int sel = (int)(tmask << (31 - sh)) >> 31;
      int v1, v2 = 0;
      if (rowx) {
        long ofs = (long)(grp ? (C0 + DELTA) : C0);
        v1 = *(const int*)(pA1 + ofs);
        if (PG) v2 = *(const int*)(pA2 + ofs);
      } else {
        v1 = *(const int*)(pA1 + C0);
        if (PG) v2 = *(const int*)(pA2 + C0);
      }
      a1[v] = v1 & sel;
      if (PG) a2[v] = v2 & sel;
    }
#pragma unroll
    for (int nt = 0; nt < NT; ++nt) {
      v8i bfr;
#pragma unroll
      for (int v = 0; v < 8; ++v) {
        const int kbB0 = kc * 64 + ((v >> 1) << 4) + ((v & 1) << 2);
        bfr[v] = wp32[wOff + nt * 16 + (kbB0 >> 2) * COUT];
      }
      const int pr = (PAR == 2) ? (kc & 1) : 0;
      acc1[nt][pr] =
          __builtin_amdgcn_wmma_i32_16x16x64_iu8(true, a1, true, bfr, acc1[nt][pr], false, false);
      if (PG)
        acc2[nt][pr] =
            __builtin_amdgcn_wmma_i32_16x16x64_iu8(true, a2, true, bfr, acc2[nt][pr], false, false);
    }
  }

#pragma unroll
  for (int nt = 0; nt < NT; ++nt) {
    v8i A1 = (PAR == 2) ? (acc1[nt][0] + acc1[nt][PAR - 1]) : acc1[nt][0];
    v8i A2 = (PAR == 2) ? (acc2[nt][0] + acc2[nt][PAR - 1]) : acc2[nt][0];
    const int n = nt * 16 + (lane & 15);
    const float s = bns[n], shb = bnb[n];
    const float t0v = PG ? th[n] : 0.0f;
#pragma unroll
    for (int v = 0; v < 8; ++v) {
      int pm = mtile * 16 + v + (grp << 3);
      float o1 = (float)A1[v];
      float val = o1;
      if (PG) {
        float o2 = 0.5f * (float)A2[v];
        val = o1 + (((o1 - t0v) > 0.0f) ? o2 : 0.0f);
      }
      float y = val * s + shb;
      long oi = (long)pm * COUT + n;
      if (HASRES) y += res[oi];
      out[oi] = y;
      if (EMIT) {
        signed char b1 = (y >= 0.0f) ? (signed char)1 : (signed char)-1;
        signed char b2 = ((y - (float)b1) >= 0.0f) ? (signed char)1 : (signed char)-1;
        ob1[oi] = b1;
        ob2[oi] = b2;
      }
    }
  }
}

__global__ void __launch_bounds__(256)
conv1_wmma(const float* __restrict__ x, const signed char* __restrict__ wp,
           const float* __restrict__ bns, const float* __restrict__ bnb,
           float* __restrict__ out, signed char* __restrict__ ob1,
           signed char* __restrict__ ob2) {
  constexpr int CIN = 96, COUT = 16, H = 32, W = 32;
  constexpr int NPIX = BATCH * H * W;
  constexpr int KTOT = CIN * 9;
  constexpr int KCH = (KTOT + 63) / 64;

  const int wave = (int)((blockIdx.x * blockDim.x + threadIdx.x) >> 5);
  const int lane = threadIdx.x & 31;
  const int mtile = wave;
  if (mtile * 16 >= NPIX) return;

  const int row = lane & 15, grp = lane >> 4;
  const int p = mtile * 16 + row;
  const int bb = p >> 10;
  const int rp = p & 1023;
  const int oy = rp >> 5, ox = rp & 31;
  const int n = lane & 15;

  const int sbase = bb * 3072 + (oy - 1) * W + (ox - 1);
  unsigned tmask = 0u;
#pragma unroll
  for (int t = 0; t < 9; ++t) {
    int ky = t / 3, kx = t - (t / 3) * 3;
    int iy = oy + ky - 1, ix = ox + kx - 1;
    unsigned bit = (iy >= 0 && iy < H && ix >= 0 && ix < W) ? 1u : 0u;
    tmask |= bit << t;
  }

  const int* wp32 = (const int*)wp;
  const int wOff = (grp << 1) * COUT + n;

  v8i acce = {}, acco = {};
#pragma unroll
  for (int kc = 0; kc < KCH; ++kc) {
    v8i a, bfr;
#pragma unroll
    for (int v = 0; v < 8; ++v) {
      const int kb0 = kc * 64 + ((v >> 1) << 4) + ((v & 1) << 2);
      const int t0 = kb0 / CIN, c0 = kb0 - t0 * CIN;
      const bool cross = (c0 + 8 >= CIN);
      const int cg0 = c0, cg1 = cross ? (c0 + 8 - CIN) : (c0 + 8);
      const int tg0 = t0, tg1 = cross ? (t0 + 1) : t0;
      const int Cg0 = (cg0 >> 5) * 1024 + (tg0 / 3) * W + (tg0 - (tg0 / 3) * 3);
      const int Cg1 = (cg1 >> 5) * 1024 + (tg1 / 3) * W + (tg1 - (tg1 / 3) * 3);
      const float tb0 = (float)(cg0 & 31) * (1.0f / (float)RESOLUTION);
      const float tb1 = (float)(cg1 & 31) * (1.0f / (float)RESOLUTION);
      int Coff = grp ? Cg1 : Cg0;
      float tb = grp ? tb1 : tb0;
      int sh = cross ? (t0 + grp) : t0;
      int sel = (int)(tmask << (31 - sh)) >> 31;
      unsigned off = (unsigned)((sbase + Coff) & sel);
      float xv = x[off];
      unsigned pk = 0u;
#pragma unroll
      for (int j = 0; j < 4; ++j) {
        float thv = tb + (((float)j + 0.5f) * (1.0f / (float)RESOLUTION));
        signed char bit = (xv > thv) ? (signed char)1 : (signed char)-1;
        pk |= ((unsigned)(unsigned char)bit) << (8 * j);
      }
      a[v] = (int)pk & sel;
      const int kbB0 = kc * 64 + ((v >> 1) << 4) + ((v & 1) << 2);
      bfr[v] = wp32[wOff + (kbB0 >> 2) * COUT];
    }
    if (kc & 1)
      acco = __builtin_amdgcn_wmma_i32_16x16x64_iu8(true, a, true, bfr, acco, false, false);
    else
      acce = __builtin_amdgcn_wmma_i32_16x16x64_iu8(true, a, true, bfr, acce, false, false);
  }
  v8i acc = acce + acco;

  const float s = bns[n], shb = bnb[n];
#pragma unroll
  for (int v = 0; v < 8; ++v) {
    int pm = mtile * 16 + v + (grp << 3);
    float y = (float)acc[v] * s + shb;
    long oi = (long)pm * COUT + n;
    out[oi] = y;
    signed char b1 = (y >= 0.0f) ? (signed char)1 : (signed char)-1;
    signed char b2 = ((y - (float)b1) >= 0.0f) ? (signed char)1 : (signed char)-1;
    ob1[oi] = b1;
    ob2[oi] = b2;
  }
}

__global__ void pool_fc_kernel(const float* __restrict__ h, const float* __restrict__ fw,
                               const float* __restrict__ fb, float* __restrict__ out) {
  __shared__ float m[64];
  int b = blockIdx.x, c = threadIdx.x;
  float sum = 0.0f;
  for (int pp = 0; pp < 64; ++pp) sum += h[((long)b * 64 + pp) * 64 + c];
  m[c] = sum * (1.0f / 64.0f);
  __syncthreads();
  if (c < 10) {
    float acc = fb[c];
    for (int j = 0; j < 64; ++j) acc += m[j] * fw[c * 64 + j];
    out[b * 10 + c] = acc;
  }
}


#define LAUNCH_CONV(CIN, COUT, STRIDE, KS, PG, HIN, WIN, EMIT, HASRES, XB1, XB2, WP, \
                    S, Bv, TH, RES, OUT, OB1, OB2)                                   \
  pg_conv_wmma<CIN, COUT, STRIDE, KS, PG, HIN, WIN, EMIT, HASRES>                    \
      <<<dim3(cdiv_h(((BATCH * ((HIN) / (STRIDE)) * ((WIN) / (STRIDE))) / 16) * 32,  \
                     256)),                                                          \
         dim3(256), 0, stream>>>(XB1, XB2, WP, S, Bv, TH, RES, OUT, OB1, OB2)

extern "C" void kernel_launch(void* const* d_in, const int* in_sizes, int n_in,
                              void* d_out, int out_size, void* d_ws, size_t ws_size,
                              hipStream_t stream) {
  (void)in_sizes; (void)n_in; (void)out_size; (void)ws_size;
  const float* x = (const float*)d_in[0];
  const float* conv1_w = (const float*)d_in[1];
  const float* bn1_s = (const float*)d_in[2];
  const float* bn1_b = (const float*)d_in[3];

  static const int cfg[9][3] = {{16, 16, 1}, {16, 16, 1}, {16, 16, 1},
                                {16, 32, 2}, {32, 32, 1}, {32, 32, 1},
                                {32, 64, 2}, {64, 64, 1}, {64, 64, 1}};
  struct BP {
    const float *wA, *thA, *sA, *bA, *wB, *thB, *sB, *bB, *wS, *sS, *bS;
  } bp[9];
  int ii = 4;
  for (int i = 0; i < 9; ++i) {
    bp[i].wA = (const float*)d_in[ii++]; bp[i].thA = (const float*)d_in[ii++];
    bp[i].sA = (const float*)d_in[ii++]; bp[i].bA  = (const float*)d_in[ii++];
    bp[i].wB = (const float*)d_in[ii++]; bp[i].thB = (const float*)d_in[ii++];
    bp[i].sB = (const float*)d_in[ii++]; bp[i].bB  = (const float*)d_in[ii++];
    bool proj = (cfg[i][2] != 1) || (cfg[i][0] != cfg[i][1]);
    if (proj) {
      bp[i].wS = (const float*)d_in[ii++];
      bp[i].sS = (const float*)d_in[ii++];
      bp[i].bS = (const float*)d_in[ii++];
    } else {
      bp[i].wS = nullptr; bp[i].sS = nullptr; bp[i].bS = nullptr;
    }
  }
  const float* fc_w = (const float*)d_in[ii++];
  const float* fc_b = (const float*)d_in[ii++];

  char* ws = (char*)d_ws;
  size_t off = 0;
  auto carve = [&](size_t bytes) -> void* {
    void* r = ws + off;
    off += (bytes + 255) & ~(size_t)255;
    return r;
  };
  const size_t GUARD = 8192;
  const size_t BITSB = (size_t)BATCH * 32 * 32 * 16;
  const size_t FBUF  = (size_t)BATCH * 32 * 32 * 16 * 4;
  auto carve_bits = [&]() -> signed char* {
    return (signed char*)carve(BITSB + 2 * GUARD) + GUARD;
  };
  signed char* bxa = carve_bits();
  signed char* bxb = carve_bits();
  signed char* bya = carve_bits();
  signed char* byb = carve_bits();
  float* f0 = (float*)carve(FBUF);
  float* f1 = (float*)carve(FBUF);
  float* f2 = (float*)carve(FBUF);

  auto kpad = [](int cin, int ks) { int kt = cin * ks * ks; return ((kt + 63) / 64) * 64; };
  signed char* wp_c1 = (signed char*)carve((size_t)kpad(96, 3) * 16);
  signed char *wpA[9], *wpB[9], *wpS[9];
  for (int i = 0; i < 9; ++i) {
    wpA[i] = (signed char*)carve((size_t)kpad(cfg[i][0], 3) * cfg[i][1]);
    wpB[i] = (signed char*)carve((size_t)kpad(cfg[i][1], 3) * cfg[i][1]);
    wpS[i] = bp[i].wS ? (signed char*)carve((size_t)kpad(cfg[i][0], 1) * cfg[i][1]) : nullptr;
  }

  auto prep = [&](const float* w, signed char* wpd, int cin, int cout, int ks) {
    int kp = kpad(cin, ks);
    int tot = kp * cout;
    prep_w_kernel<<<dim3(cdiv_h(tot, 256)), dim3(256), 0, stream>>>(w, wpd, cin, cout, ks, kp);
  };
  prep(conv1_w, wp_c1, 96, 16, 3);
  for (int i = 0; i < 9; ++i) {
    prep(bp[i].wA, wpA[i], cfg[i][0], cfg[i][1], 3);
    prep(bp[i].wB, wpB[i], cfg[i][1], cfg[i][1], 3);
    if (bp[i].wS) prep(bp[i].wS, wpS[i], cfg[i][0], cfg[i][1], 1);
  }

  {
    int waves = (BATCH * 32 * 32) / 16;
    conv1_wmma<<<dim3(cdiv_h(waves * 32, 256)), dim3(256), 0, stream>>>(
        x, wp_c1, bn1_s, bn1_b, f0, bxa, bxb);
  }

  float *X = f0, *T1 = f1, *T2 = f2;

  for (int i = 0; i < 3; ++i) {
    LAUNCH_CONV(16, 16, 1, 3, true, 32, 32, true, false, bxa, bxb, wpA[i],
                bp[i].sA, bp[i].bA, bp[i].thA, (const float*)nullptr, T1, bya, byb);
    LAUNCH_CONV(16, 16, 1, 3, true, 32, 32, true, true, bya, byb, wpB[i],
                bp[i].sB, bp[i].bB, bp[i].thB, X, T2, bxa, bxb);
    float* tmp = X; X = T2; T2 = tmp;
  }

  LAUNCH_CONV(16, 32, 2, 1, false, 32, 32, false, false, bxa, (const signed char*)nullptr,
              wpS[3], bp[3].sS, bp[3].bS, (const float*)nullptr, (const float*)nullptr,
              T2, (signed char*)nullptr, (signed char*)nullptr);
  LAUNCH_CONV(16, 32, 2, 3, true, 32, 32, true, false, bxa, bxb, wpA[3],
              bp[3].sA, bp[3].bA, bp[3].thA, (const float*)nullptr, T1, bya, byb);
  LAUNCH_CONV(32, 32, 1, 3, true, 16, 16, true, true, bya, byb, wpB[3],
              bp[3].sB, bp[3].bB, bp[3].thB, T2, X, bxa, bxb);

  for (int i = 4; i < 6; ++i) {
    LAUNCH_CONV(32, 32, 1, 3, true, 16, 16, true, false, bxa, bxb, wpA[i],
                bp[i].sA, bp[i].bA, bp[i].thA, (const float*)nullptr, T1, bya, byb);
    LAUNCH_CONV(32, 32, 1, 3, true, 16, 16, true, true, bya, byb, wpB[i],
                bp[i].sB, bp[i].bB, bp[i].thB, X, T2, bxa, bxb);
    float* tmp = X; X = T2; T2 = tmp;
  }

  LAUNCH_CONV(32, 64, 2, 1, false, 16, 16, false, false, bxa, (const signed char*)nullptr,
              wpS[6], bp[6].sS, bp[6].bS, (const float*)nullptr, (const float*)nullptr,
              T2, (signed char*)nullptr, (signed char*)nullptr);
  LAUNCH_CONV(32, 64, 2, 3, true, 16, 16, true, false, bxa, bxb, wpA[6],
              bp[6].sA, bp[6].bA, bp[6].thA, (const float*)nullptr, T1, bya, byb);
  LAUNCH_CONV(64, 64, 1, 3, true, 8, 8, true, true, bya, byb, wpB[6],
              bp[6].sB, bp[6].bB, bp[6].thB, T2, X, bxa, bxb);

  for (int i = 7; i < 9; ++i) {
    LAUNCH_CONV(64, 64, 1, 3, true, 8, 8, true, false, bxa, bxb, wpA[i],
                bp[i].sA, bp[i].bA, bp[i].thA, (const float*)nullptr, T1, bya, byb);
    LAUNCH_CONV(64, 64, 1, 3, true, 8, 8, true, true, bya, byb, wpB[i],
                bp[i].sB, bp[i].bB, bp[i].thB, X, T2, bxa, bxb);
    float* tmp = X; X = T2; T2 = tmp;
  }

  pool_fc_kernel<<<dim3(BATCH), dim3(64), 0, stream>>>(X, fc_w, fc_b, (float*)d_out);
}
